// ParticleDecoder_21827023798294
// MI455X (gfx1250) — matrix operand order repaired and checked by reading, not compiled, not run on hardware
//
#include <hip/hip_runtime.h>

typedef __attribute__((ext_vector_type(16))) _Float16 v16h;
typedef __attribute__((ext_vector_type(2)))  _Float16 v2h;
typedef __attribute__((ext_vector_type(8)))  float    v8f;

#define ROWS       32768
#define IN_PAD     64
#define HDIM       128
#define OUTC       48
#define LDS_STRIDE 136   // f16 elements per staged row (272B => rows offset 4 banks)

enum { ACT_NONE = 0, ACT_RELU = 1, ACT_TANH = 2 };

__device__ __forceinline__ void lds_fence() {
  asm volatile("s_wait_dscnt 0" ::: "memory");
}

// A-matrix fragment (16x32 f16) from LDS stage, per CDNA5 ISA 7.12.2:
// lanes 0-15: M=lane, K pairs {0,1,2,3,4,5,6,7 ...} base 0 ; lanes 16-31: base K=8.
// VGPR i<4 -> K = kb+2i ; VGPR i>=4 -> K = kb+16+2(i-4).
__device__ __forceinline__ v16h load_A(const _Float16* s, int lane, int kstep) {
  const int m  = lane & 15;
  const int kb = kstep * 32 + ((lane & 16) ? 8 : 0);
  v16h a;
#pragma unroll
  for (int i = 0; i < 8; ++i) {
    const int k = kb + ((i < 4) ? (2 * i) : (16 + 2 * (i - 4)));
    v2h p = *(const v2h*)(s + m * LDS_STRIDE + k);
    a[2 * i]     = p.x;
    a[2 * i + 1] = p.y;
  }
  return a;
}

// B fragment: weights are pre-swizzled so each lane's 16 f16 are contiguous.
__device__ __forceinline__ v16h load_B(const _Float16* w, int ntile, int ksteps,
                                       int kstep, int lane) {
  return *(const v16h*)(w + ((size_t)(ntile * ksteps + kstep) * 512 + lane * 16));
}

__device__ __forceinline__ void mlp_layer(const _Float16* in_s, _Float16* out_s,
                                          const _Float16* w, const float* bias,
                                          int ksteps, int ntiles, int act, int lane) {
  for (int nt = 0; nt < ntiles; ++nt) {
    const int n = nt * 16 + (lane & 15);
    const float b = bias[n];
    v8f acc = {b, b, b, b, b, b, b, b};
    for (int ks = 0; ks < ksteps; ++ks) {
      v16h a  = load_A(in_s, lane, ks);
      v16h bm = load_B(w, nt, ksteps, ks, lane);
      acc = __builtin_amdgcn_wmma_f32_16x16x32_f16(false, a, false, bm,
                                                   (short)0, acc, false, false);
    }
#pragma unroll
    for (int v = 0; v < 8; ++v) {
      float x = acc[v];
      if (act == ACT_RELU)      x = fmaxf(x, 0.0f);
      else if (act == ACT_TANH) x = tanhf(x);
      const int row = ((lane & 16) ? 8 : 0) + v;   // C/D layout: VGPR v -> M=v or v+8
      out_s[row * LDS_STRIDE + n] = (_Float16)x;
    }
  }
  lds_fence();
}

// ---------------- weight pack: fp32 (K,N) row-major -> f16 WMMA-B fragment order
// dst index e = ((ntile*ksteps + kstep)*32 + lane)*16 + j ; value = W[k][n],
// k = kstep*32 + (lane>=16 ? 8 : 0) + j + (j>=8 ? 8 : 0) (matches A table order), n = ntile*16 + lane&15.
__global__ void pack_b_kernel(const float* __restrict__ src, _Float16* __restrict__ dst,
                              int Ksrc, int N, int ksteps, int ntiles, int mask_mode) {
  const int e = blockIdx.x * blockDim.x + threadIdx.x;
  const int total = ntiles * ksteps * 512;
  if (e >= total) return;
  const int j     = e & 15;
  const int lane  = (e >> 4) & 31;
  const int kstep = (e >> 9) % ksteps;
  const int ntile = (e >> 9) / ksteps;
  const int k = kstep * 32 + ((lane & 16) ? 8 : 0) + j + ((j >= 8) ? 8 : 0);
  const int n = ntile * 16 + (lane & 15);
  float v = (k < Ksrc) ? src[k * N + n] : 0.0f;
  if (mask_mode == 1) {               // m2: keep if deg_h[n] >= deg_h[k]
    if ((n % 7 + 1) < (k % 7 + 1)) v = 0.0f;
  } else if (mask_mode == 2) {        // mo: keep if deg_out[n] > deg_h[k]
    if (!((n / 6 + 1) > (k % 7 + 1))) v = 0.0f;
  }
  dst[e] = (_Float16)v;
}

// ---------------- KNN: per-batch all-pairs in LDS, register sort network
__global__ __launch_bounds__(256) void knn_kernel(const float* __restrict__ coords,
                                                  _Float16* __restrict__ xh) {
  __shared__ float cx[1024], cy[1024], cz[1024];
  const int b    = blockIdx.y;
  const int base = b * 1024;
  for (int t = threadIdx.x; t < 1024; t += 256) {
    cx[t] = coords[(size_t)(base + t) * 3 + 0];
    cy[t] = coords[(size_t)(base + t) * 3 + 1];
    cz[t] = coords[(size_t)(base + t) * 3 + 2];
  }
  __syncthreads();
  const int i = blockIdx.x * 256 + threadIdx.x;
  const float px = cx[i], py = cy[i], pz = cz[i];

  float bd[13];
  int   bi[13];
#pragma unroll
  for (int t = 0; t < 13; ++t) { bd[t] = 3.0e38f; bi[t] = 0x7fffffff; }

  for (int j = 0; j < 1024; ++j) {
    const float dx = cx[j] - px, dy = cy[j] - py, dz = cz[j] - pz;
    const float d = dx * dx + dy * dy + dz * dz;
    if (d < bd[12] || (d == bd[12] && j < bi[12])) {
      float cd = d; int cj = j;
#pragma unroll
      for (int t = 0; t < 13; ++t) {   // stable insertion (value, then index)
        const bool better = (cd < bd[t]) || (cd == bd[t] && cj < bi[t]);
        const float td = bd[t]; const int tj = bi[t];
        if (better) { bd[t] = cd; bi[t] = cj; cd = td; cj = tj; }
      }
    }
  }
  const size_t row = (size_t)base + i;
  _Float16* xr = xh + row * IN_PAD;
#pragma unroll
  for (int t = 1; t <= 12; ++t) {      // drop slot 0 (self), like idx[:, :, 1:]
    const int nb = bi[t];
    xr[(t - 1) * 3 + 0] = (_Float16)(cx[nb] - px);
    xr[(t - 1) * 3 + 1] = (_Float16)(cy[nb] - py);
    xr[(t - 1) * 3 + 2] = (_Float16)(cz[nb] - pz);
  }
#pragma unroll
  for (int c = 36; c < IN_PAD; ++c) xr[c] = (_Float16)0.0f;
}

// ---------------- fused 7-GEMM MLP: one wave per 16-row tile
__global__ __launch_bounds__(32) void mlp_wmma_kernel(
    const _Float16* __restrict__ xh, const float* __restrict__ coords,
    const _Float16* w1, const float* b1, const _Float16* w2, const float* b2,
    const _Float16* w3, const float* b3, const _Float16* wbp, const float* bbp,
    const _Float16* cw, const float* mb1, const _Float16* mw2, const float* mb2,
    const _Float16* mwo, const float* mbo, float* __restrict__ out) {
  __shared__ _Float16 smem[3 * 16 * LDS_STRIDE];
  _Float16* S0 = smem;
  _Float16* S1 = smem + 16 * LDS_STRIDE;
  _Float16* S2 = smem + 32 * LDS_STRIDE;
  const int lane = threadIdx.x;
  const int r0   = blockIdx.x * 16;

  {  // stage 16 rows x 64 f16 of input, one dword per lane per row
    const unsigned int* xs = (const unsigned int*)(xh + (size_t)r0 * IN_PAD);
#pragma unroll
    for (int row = 0; row < 16; ++row) {
      const unsigned int u = xs[row * (IN_PAD / 2) + lane];
      *(unsigned int*)(S0 + row * LDS_STRIDE + lane * 2) = u;
    }
  }
  lds_fence();

  mlp_layer(S0, S1, w1,  b1,  2, 8, ACT_RELU, lane);  // h1 = relu(x @ W1)
  mlp_layer(S0, S2, cw,  mb1, 2, 8, ACT_TANH, lane);  // m1 = tanh(x @ cw)  (x0 term is 0)
  mlp_layer(S1, S0, w2,  b2,  4, 8, ACT_RELU, lane);  // h2
  mlp_layer(S0, S1, w3,  b3,  4, 8, ACT_RELU, lane);  // h3 -> S1
  mlp_layer(S2, S0, mw2, mb2, 4, 8, ACT_TANH, lane);  // m2 -> S0

  // final: params = h3@wbp + b ; shifts = m2@mwo + b ; fuse combine + exp + store
  for (int nt = 0; nt < 3; ++nt) {
    const int n = nt * 16 + (lane & 15);
    const float pb = bbp[n], sb = mbo[n];
    v8f aP = {pb, pb, pb, pb, pb, pb, pb, pb};
    v8f aS = {sb, sb, sb, sb, sb, sb, sb, sb};
    for (int ks = 0; ks < 4; ++ks) {
      v16h a1 = load_A(S1, lane, ks);
      v16h f1 = load_B(wbp, nt, 4, ks, lane);
      aP = __builtin_amdgcn_wmma_f32_16x16x32_f16(false, a1, false, f1,
                                                  (short)0, aP, false, false);
      v16h a2 = load_A(S0, lane, ks);
      v16h f2 = load_B(mwo, nt, 4, ks, lane);
      aS = __builtin_amdgcn_wmma_f32_16x16x32_f16(false, a2, false, f2,
                                                  (short)0, aS, false, false);
    }
    const int isStd = n & 1;          // flat col c = s*6 + d*2 + p  (p = c&1)
    const int d     = (n >> 1) % 3;
#pragma unroll
    for (int v = 0; v < 8; ++v) {
      const int row = r0 + ((lane & 16) ? 8 : 0) + v;
      const float val = aP[v] + aS[v];
      const float o = isStd ? expf(0.5f * val) : (coords[row * 3 + d] + val);
      out[(size_t)row * OUTC + n] = o;
    }
  }
}

extern "C" void kernel_launch(void* const* d_in, const int* in_sizes, int n_in,
                              void* d_out, int out_size, void* d_ws, size_t ws_size,
                              hipStream_t stream) {
  (void)in_sizes; (void)n_in; (void)out_size; (void)ws_size;
  const float* coords = (const float*)d_in[0];
  const float* w_h1 = (const float*)d_in[1];
  const float* b_h1 = (const float*)d_in[2];
  const float* w_h2 = (const float*)d_in[3];
  const float* b_h2 = (const float*)d_in[4];
  const float* w_h3 = (const float*)d_in[5];
  const float* b_h3 = (const float*)d_in[6];
  const float* w_bp = (const float*)d_in[7];
  const float* b_bp = (const float*)d_in[8];
  const float* cw   = (const float*)d_in[9];
  // d_in[10] = mw1 is dead (x0 == 0)
  const float* mb1  = (const float*)d_in[11];
  const float* mw2  = (const float*)d_in[12];
  const float* mb2  = (const float*)d_in[13];
  const float* mwo  = (const float*)d_in[14];
  const float* mbo  = (const float*)d_in[15];
  float* out = (float*)d_out;

  char* ws = (char*)d_ws;
  _Float16* xh = (_Float16*)ws;
  size_t off = (size_t)ROWS * IN_PAD * 2;                       // 4 MB
  _Float16* w1f  = (_Float16*)(ws + off); off += 64  * 128 * 2;
  _Float16* w2f  = (_Float16*)(ws + off); off += 128 * 128 * 2;
  _Float16* w3f  = (_Float16*)(ws + off); off += 128 * 128 * 2;
  _Float16* wbpf = (_Float16*)(ws + off); off += 128 * 48  * 2;
  _Float16* cwf  = (_Float16*)(ws + off); off += 64  * 128 * 2;
  _Float16* mw2f = (_Float16*)(ws + off); off += 128 * 128 * 2;
  _Float16* mwof = (_Float16*)(ws + off); off += 128 * 48  * 2;

  pack_b_kernel<<<8192  / 256, 256, 0, stream>>>(w_h1, w1f,  36,  128, 2, 8, 0);
  pack_b_kernel<<<16384 / 256, 256, 0, stream>>>(w_h2, w2f,  128, 128, 4, 8, 0);
  pack_b_kernel<<<16384 / 256, 256, 0, stream>>>(w_h3, w3f,  128, 128, 4, 8, 0);
  pack_b_kernel<<<6144  / 256, 256, 0, stream>>>(w_bp, wbpf, 128, 48,  4, 3, 0);
  pack_b_kernel<<<8192  / 256, 256, 0, stream>>>(cw,   cwf,  36,  128, 2, 8, 0);
  pack_b_kernel<<<16384 / 256, 256, 0, stream>>>(mw2,  mw2f, 128, 128, 4, 8, 1);
  pack_b_kernel<<<6144  / 256, 256, 0, stream>>>(mwo,  mwof, 128, 48,  4, 3, 2);

  knn_kernel<<<dim3(4, 32), 256, 0, stream>>>(coords, xh);

  mlp_wmma_kernel<<<ROWS / 16, 32, 0, stream>>>(xh, coords,
      w1f, b_h1, w2f, b_h2, w3f, b_h3, wbpf, b_bp,
      cwf, mb1, mw2f, mb2, mwof, mbo, out);
}
